// HashMLPModel_74629351735872
// MI455X (gfx1250) — matrix operand order repaired and checked by reading, not compiled, not run on hardware
//
#include <hip/hip_runtime.h>

typedef __attribute__((ext_vector_type(16))) _Float16 v16h;
typedef __attribute__((ext_vector_type(8)))  _Float16 v8h;
typedef __attribute__((ext_vector_type(8)))  float    v8f;

#define NUM_LEVELS 32
#define LOG2_T     19
#define TBL        (1u << LOG2_T)
#define TMASK      (TBL - 1u)

// Packed fp16 weight-fragment layout in d_ws (offsets in fragments of 512 _Float16).
// Fragment = 32 lanes x 16 f16 = 512 elements. Lane data contiguous (32B).
#define W1_FRAG0   0      // 16 frags (kt 0..1, nt 0..7)
#define W2_FRAG0   16     // 32 frags (kt 0..3, nt 0..7)
#define W3_FRAG0   48     // 4 frags  (kt 0..3, nt=0, col0 = W3, rest 0)
#define WP_TOTAL   (52 * 512)

__global__ void pack_weights_kernel(const float* __restrict__ W1,
                                    const float* __restrict__ W2,
                                    const float* __restrict__ W3,
                                    _Float16* __restrict__ Wp)
{
    int idx = blockIdx.x * blockDim.x + threadIdx.x;
    if (idx >= WP_TOTAL) return;
    int f = idx >> 9;           // global fragment index
    int r = idx & 511;
    int L = r >> 4;             // lane 0..31
    int j = r & 15;             // K-within-lane 0..15
    int col16 = L & 15;
    int khalf = L >> 4;
    float v;
    if (f < W2_FRAG0) {                         // layer 1: [64 x 128]
        int kt = f >> 3, nt = f & 7;
        int kb = kt * 32 + khalf * 8 + (j & 8);
        v = W1[(kb + j) * 128 + nt * 16 + col16];
    } else if (f < W3_FRAG0) {                  // layer 2: [128 x 128]
        int f2 = f - W2_FRAG0;
        int kt = f2 >> 3, nt = f2 & 7;
        int kb = kt * 32 + khalf * 8 + (j & 8);
        v = W2[(kb + j) * 128 + nt * 16 + col16];
    } else {                                    // layer 3: [128 x 1] padded to [128 x 16]
        int kt = f - W3_FRAG0;
        int kb = kt * 32 + khalf * 8 + (j & 8);
        v = (col16 == 0) ? W3[kb + j] : 0.0f;
    }
    Wp[idx] = (_Float16)v;
}

__device__ __forceinline__ v16h wmma_f16(v16h a, v16h b, v8f& c) {
    c = __builtin_amdgcn_wmma_f32_16x16x32_f16(
        /*neg_a=*/false, a, /*neg_b=*/false, b,
        /*c_mod=*/(short)0, c, /*reuse_a=*/false, /*reuse_b=*/false);
    return a;
}

__launch_bounds__(128)
__global__ void hashmlp_fused_kernel(const float*  __restrict__ x,
                                     const float2* __restrict__ emb,
                                     const float*  __restrict__ b1,
                                     const float*  __restrict__ b2,
                                     const float*  __restrict__ b3,
                                     const _Float16* __restrict__ Wp,
                                     float* __restrict__ out)
{
    __shared__ float    sX[64 * 3];
    __shared__ _Float16 sH0[64 * 64];    // hash features  (stride 64)
    __shared__ _Float16 sH1[64 * 128];   // layer-1 output (stride 128)
    __shared__ _Float16 sH2[64 * 128];   // layer-2 output (stride 128)

    const int tid  = threadIdx.x;
    const int lane = tid & 31;
    const int wave = tid >> 5;
    const int p0   = blockIdx.x * 64;

    __builtin_prefetch((const void*)Wp, 0, 0);   // global_prefetch_b8 on hot weights

    // ---- stage coordinates ----
    if (tid < 192) sX[tid] = x[(size_t)p0 * 3 + tid];
    __syncthreads();

    // ---- phase 1: multi-resolution hash encode -> sH0 [64 x 64] f16 ----
    for (int i = 0; i < 16; ++i) {
        int task = tid + 128 * i;       // 0..2047 bijective
        int p = task & 63;
        int l = task >> 6;              // 0..31
        float xx = sX[p * 3 + 0], yy = sX[p * 3 + 1], zz = sX[p * 3 + 2];
        unsigned res = 1u << l;
        float scale = (float)res - 1.0f;
        float px = xx * scale + 0.5f;
        float py = yy * scale + 0.5f;
        float pz = zz * scale + 0.5f;
        float x0 = floorf(px), y0 = floorf(py), z0 = floorf(pz);
        float fx = px - x0, fy = py - y0, fz = pz - z0;
        unsigned gx = (unsigned)x0, gy = (unsigned)y0, gz = (unsigned)z0;
        bool dense = (l <= 6);          // res^3 <= 2^19
        const float2* tab = emb + (size_t)l * TBL;
        float f0 = 0.0f, f1 = 0.0f;
        #pragma unroll
        for (int c = 0; c < 8; ++c) {
            unsigned cx = gx + (c & 1);
            unsigned cy = gy + ((c >> 1) & 1);
            unsigned cz = gz + ((c >> 2) & 1);
            unsigned idx = dense
                ? (cx + cy * res + cz * res * res)
                : (cx ^ (cy * 2654435761u) ^ (cz * 805459861u));
            idx &= TMASK;
            float2 e = tab[idx];
            float w = ((c & 1)        ? fx : 1.0f - fx)
                    * (((c >> 1) & 1) ? fy : 1.0f - fy)
                    * (((c >> 2) & 1) ? fz : 1.0f - fz);
            f0 += w * e.x;
            f1 += w * e.y;
        }
        union { _Float16 h[2]; unsigned u; } pk;
        pk.h[0] = (_Float16)f0;
        pk.h[1] = (_Float16)f1;
        *(unsigned*)&sH0[p * 64 + 2 * l] = pk.u;   // ds_store_b32
    }
    __syncthreads();

    const int m    = lane & 15;   // row within tile / output column
    const int kh   = lane >> 4;   // K-half selector
    const int rowB = wave * 16;   // 16 rows per wave

    // A fragment: ISA 16-bit A 16x32 layout -> two ds_load_b128 chunks
    auto loadA = [&](const _Float16* h, int stride, int kt) -> v16h {
        const _Float16* row = h + (rowB + m) * stride + kt * 32 + kh * 8;
        v16h a;
        ((v8h*)&a)[0] = *(const v8h*)(row);        // K = kt*32 + kh*8 .. +8
        ((v8h*)&a)[1] = *(const v8h*)(row + 16);   // K = kt*32 + 16 + kh*8 .. +8
        return a;
    };
    auto loadB = [&](int frag) -> v16h {           // one contiguous 32B global load
        return *(const v16h*)(Wp + ((size_t)frag * 32 + lane) * 16);
    };

    // ---- layer 1: [64x64] @ [64x128], bias + ReLU -> sH1 ----
    {
        v16h a0 = loadA(sH0, 64, 0);
        v16h a1 = loadA(sH0, 64, 1);
        #pragma unroll
        for (int nt = 0; nt < 8; ++nt) {
            v8f c = {};
            wmma_f16(a0, loadB(W1_FRAG0 + 0 * 8 + nt), c);
            wmma_f16(a1, loadB(W1_FRAG0 + 1 * 8 + nt), c);
            float bias = b1[nt * 16 + m];
            #pragma unroll
            for (int v = 0; v < 8; ++v) {
                float r = c[v] + bias;
                r = fmaxf(r, 0.0f);
                sH1[(rowB + v + kh * 8) * 128 + nt * 16 + m] = (_Float16)r;
            }
        }
    }
    __syncthreads();

    // ---- layer 2: [64x128] @ [128x128], bias + ReLU -> sH2 ----
    {
        v16h a0 = loadA(sH1, 128, 0);
        v16h a1 = loadA(sH1, 128, 1);
        v16h a2 = loadA(sH1, 128, 2);
        v16h a3 = loadA(sH1, 128, 3);
        #pragma unroll
        for (int nt = 0; nt < 8; ++nt) {
            v8f c = {};
            wmma_f16(a0, loadB(W2_FRAG0 + 0 * 8 + nt), c);
            wmma_f16(a1, loadB(W2_FRAG0 + 1 * 8 + nt), c);
            wmma_f16(a2, loadB(W2_FRAG0 + 2 * 8 + nt), c);
            wmma_f16(a3, loadB(W2_FRAG0 + 3 * 8 + nt), c);
            float bias = b2[nt * 16 + m];
            #pragma unroll
            for (int v = 0; v < 8; ++v) {
                float r = c[v] + bias;
                r = fmaxf(r, 0.0f);
                sH2[(rowB + v + kh * 8) * 128 + nt * 16 + m] = (_Float16)r;
            }
        }
    }
    __syncthreads();

    // ---- layer 3: [64x128] @ [128x16(pad)], only column 0 valid ----
    {
        v16h a0 = loadA(sH2, 128, 0);
        v16h a1 = loadA(sH2, 128, 1);
        v16h a2 = loadA(sH2, 128, 2);
        v16h a3 = loadA(sH2, 128, 3);
        v8f c = {};
        wmma_f16(a0, loadB(W3_FRAG0 + 0), c);
        wmma_f16(a1, loadB(W3_FRAG0 + 1), c);
        wmma_f16(a2, loadB(W3_FRAG0 + 2), c);
        wmma_f16(a3, loadB(W3_FRAG0 + 3), c);
        float bv = b3[0];
        if (m == 0) {   // lanes 0 and 16 hold output column N=0
            #pragma unroll
            for (int v = 0; v < 8; ++v)
                out[p0 + rowB + v + kh * 8] = c[v] + bv;
        }
    }
}

extern "C" void kernel_launch(void* const* d_in, const int* in_sizes, int n_in,
                              void* d_out, int out_size, void* d_ws, size_t ws_size,
                              hipStream_t stream) {
    const float*  x   = (const float*)d_in[0];
    const float2* emb = (const float2*)d_in[1];
    const float*  W1  = (const float*)d_in[2];
    const float*  b1  = (const float*)d_in[3];
    const float*  W2  = (const float*)d_in[4];
    const float*  b2  = (const float*)d_in[5];
    const float*  W3  = (const float*)d_in[6];
    const float*  b3  = (const float*)d_in[7];
    float* out = (float*)d_out;
    _Float16* Wp = (_Float16*)d_ws;

    int n = in_sizes[0] / 3;   // number of points (1<<20)

    pack_weights_kernel<<<(WP_TOTAL + 255) / 256, 256, 0, stream>>>(W1, W2, W3, Wp);

    int blocks = (n + 63) / 64;
    hashmlp_fused_kernel<<<blocks, 128, 0, stream>>>(x, emb, b1, b2, b3, Wp, out);
}
